// XMRCA_5557687681441
// MI455X (gfx1250) — matrix operand order repaired and checked by reading, not compiled, not run on hardware
//
#include <hip/hip_runtime.h>

typedef _Float16 h16v __attribute__((ext_vector_type(16)));
typedef _Float16 h8v  __attribute__((ext_vector_type(8)));
typedef _Float16 h4v  __attribute__((ext_vector_type(4)));
typedef float    f8v  __attribute__((ext_vector_type(8)));
typedef float    f4v  __attribute__((ext_vector_type(4)));
typedef float    f2v  __attribute__((ext_vector_type(2)));

#define WAVES_PER_BLOCK 8
#define BLOCK_THREADS   (WAVES_PER_BLOCK * 32)
#define SMEM_WEIGHTS    10560                       // 10240B weights + 288B params, padded/aligned
#define SMEM_PER_WAVE   18432                       // 5x 2KB f16 bufs + 2x 4KB f32 bufs
#define SMEM_TOTAL      (SMEM_WEIGHTS + WAVES_PER_BLOCK * SMEM_PER_WAVE)

// ---- cross-lane helpers without LDS traffic ----
// DPP quad_perm / row_xmask lane-xor within 16-lane rows (VALU only)
template <int CTRL>
static __device__ __forceinline__ float fxor_dpp(float v) {
  int s = __builtin_bit_cast(int, v);
  return __builtin_bit_cast(float,
      __builtin_amdgcn_update_dpp(s, s, CTRL, 0xF, 0xF, true));
}
#define DPP_XOR1 0xB1   // quad_perm [1,0,3,2]
#define DPP_XOR2 0x4E   // quad_perm [2,3,0,1]
#define DPP_XOR4 0x164  // ROW_XMASK 4
#define DPP_XOR8 0x168  // ROW_XMASK 8
// lane ^ 16 via v_permlanex16 (identity selects swap the 16-lane halves)
static __device__ __forceinline__ float fxor16(float v) {
  int s = __builtin_bit_cast(int, v);
  return __builtin_bit_cast(float,
      __builtin_amdgcn_permlanex16(s, s, 0x76543210, 0xfedcba98, true, false));
}

// A operand (f16 16x32): lanes 0-15 row M, halves = K{0..7,16..23}; lanes 16-31 K{8..15,24..31}
static __device__ __forceinline__ h16v loadA16(const _Float16* m, int row, int lane) {
  const int kOff = (lane & 16) ? 8 : 0;
  h8v lo = *(const h8v*)(m + row * 32 + kOff);
  h8v hi = *(const h8v*)(m + row * 32 + kOff + 16);
  h16v a;
#pragma unroll
  for (int t = 0; t < 8; ++t) { a[t] = lo[t]; a[t + 8] = hi[t]; }
  return a;
}

// B operand (f16 32x16): lane n holds column, K{0..7,16..23} (lanes<16) / K{8..15,24..31} (lanes>=16).
static __device__ __forceinline__ h16v loadB16(const _Float16* m, int colBase, int lane) {
  const int kOff = (lane & 16) ? 8 : 0;
  const _Float16* p = m + (colBase + (lane & 15)) * 32 + kOff;
  h8v lo = *(const h8v*)(p);
  h8v hi = *(const h8v*)(p + 16);
  h16v b;
#pragma unroll
  for (int t = 0; t < 8; ++t) { b[t] = lo[t]; b[t + 8] = hi[t]; }
  return b;
}

static __device__ __forceinline__ f8v wmma_f16(h16v a, h16v b, f8v c) {
  return __builtin_amdgcn_wmma_f32_16x16x32_f16(false, a, false, b, (short)0, c, false, false);
}
static __device__ __forceinline__ f8v wmma_f32k4(f2v a, f2v b, f8v c) {
  return __builtin_amdgcn_wmma_f32_16x16x4_f32(false, a, false, b, (short)0, c, false, false);
}

__global__ __launch_bounds__(BLOCK_THREADS)
void XMRCA_attn_ln_kernel(const float* __restrict__ x, const float* __restrict__ y,
                          const float* __restrict__ Wq, const float* __restrict__ Wkv,
                          const float* __restrict__ Wo, const float* __restrict__ Wr,
                          const float* __restrict__ temp, const float* __restrict__ gamma,
                          const float* __restrict__ beta, float* __restrict__ out, int nb) {
  extern __shared__ char smem[];
  _Float16* sWqT  = (_Float16*)smem;      // [32 out][32 in]
  _Float16* sWkvT = sWqT + 1024;          // [64 out][32 in]
  _Float16* sWoT  = sWkvT + 2048;         // [32 out][32 in]
  _Float16* sWrT  = sWoT + 1024;          // [32 out][32 in]
  float* sGamma = (float*)(sWrT + 1024);  // 32
  float* sBeta  = sGamma + 32;            // 32
  float* sTemp  = sBeta + 32;             // 8

  const int tid = threadIdx.x;
  // Stage weights once per block, transposed to [out][in], converted to f16.
  for (int i = tid; i < 1024; i += BLOCK_THREADS) {
    int r = i >> 5, c = i & 31;
    sWqT[c * 32 + r] = (_Float16)Wq[i];
    sWoT[c * 32 + r] = (_Float16)Wo[i];
    sWrT[c * 32 + r] = (_Float16)Wr[i];
  }
  for (int i = tid; i < 2048; i += BLOCK_THREADS) {
    int r = i >> 6, c = i & 63;
    sWkvT[c * 32 + r] = (_Float16)Wkv[i];
  }
  if (tid < 32) { sGamma[tid] = gamma[tid]; sBeta[tid] = beta[tid]; }
  if (tid < 8)  { sTemp[tid] = temp[tid]; }
  __syncthreads();

  const int lane = tid & 31;
  const int wave = tid >> 5;
  const int e = blockIdx.x * WAVES_PER_BLOCK + wave;
  if (e >= nb) return;

  char* wb = smem + SMEM_WEIGHTS + wave * SMEM_PER_WAVE;
  _Float16* sX  = (_Float16*)wb;   // [l][c] f16
  _Float16* sY  = sX + 1024;       // [l][c] f16
  _Float16* sVT = sY + 1024;       // [c][l] f16 (v transposed)
  _Float16* sP  = sVT + 1024;      // [l][m] f16 (probs, per-head reuse)
  _Float16* sO  = sP + 1024;       // [l][c] f16 (attention output)
  float* sQ = (float*)(sO + 1024); // [l][c] f32 (normalized q)
  float* sK = sQ + 1024;           // [l][c] f32 (normalized k)

  const int lo16 = lane & 15;
  const int hi16 = (lane & 16) ? 8 : 0;   // D-tile row offset for upper half-wave
  const f8v zero = {};

  // ---- load x,y rows (coalesced b128, nontemporal: streamed once), convert to f16 into LDS
  {
    const float* xr = x + (size_t)e * 1024 + lane * 32;
    const float* yr = y + (size_t)e * 1024 + lane * 32;
#pragma unroll
    for (int c0 = 0; c0 < 32; c0 += 8) {
      f4v a0 = __builtin_nontemporal_load((const f4v*)(xr + c0));
      f4v a1 = __builtin_nontemporal_load((const f4v*)(xr + c0 + 4));
      f4v b0 = __builtin_nontemporal_load((const f4v*)(yr + c0));
      f4v b1 = __builtin_nontemporal_load((const f4v*)(yr + c0 + 4));
      h8v hx, hy;
#pragma unroll
      for (int t = 0; t < 4; ++t) {
        hx[t] = (_Float16)a0[t]; hx[t + 4] = (_Float16)a1[t];
        hy[t] = (_Float16)b0[t]; hy[t + 4] = (_Float16)b1[t];
      }
      *(h8v*)(sX + lane * 32 + c0) = hx;
      *(h8v*)(sY + lane * 32 + c0) = hy;
    }
  }

  // ---- projections: q = x@Wq, k|v = y@Wkv   (12x v_wmma_f32_16x16x32_f16)
  f8v qd[2][2], kd[2][2], vd[2][2];
#pragma unroll
  for (int i = 0; i < 2; ++i) {
    h16v axi = loadA16(sX, 16 * i + lo16, lane);
    h16v ayi = loadA16(sY, 16 * i + lo16, lane);
#pragma unroll
    for (int j = 0; j < 2; ++j) {
      qd[i][j] = wmma_f16(axi, loadB16(sWqT, 16 * j, lane), zero);
      kd[i][j] = wmma_f16(ayi, loadB16(sWkvT, 16 * j, lane), zero);
      vd[i][j] = wmma_f16(ayi, loadB16(sWkvT, 32 + 16 * j, lane), zero);
    }
  }

  // ---- l2-normalize q,k per (row, head): head = 4 adjacent columns = 4 adjacent lanes (DPP xor)
#pragma unroll
  for (int i = 0; i < 2; ++i)
#pragma unroll
    for (int j = 0; j < 2; ++j)
#pragma unroll
      for (int r = 0; r < 8; ++r) {
        float qv = qd[i][j][r], kv = kd[i][j][r];
        float sq = qv * qv; sq += fxor_dpp<DPP_XOR1>(sq); sq += fxor_dpp<DPP_XOR2>(sq);
        float sk = kv * kv; sk += fxor_dpp<DPP_XOR1>(sk); sk += fxor_dpp<DPP_XOR2>(sk);
        qd[i][j][r] = qv * rsqrtf(fmaxf(sq, 1e-24f));
        kd[i][j][r] = kv * rsqrtf(fmaxf(sk, 1e-24f));
      }

  // ---- stash q,k (f32 row-major) and v (f16 transposed, contiguous b128 stores)
#pragma unroll
  for (int i = 0; i < 2; ++i)
#pragma unroll
    for (int j = 0; j < 2; ++j) {
      h8v hv;
#pragma unroll
      for (int r = 0; r < 8; ++r) {
        int row = 16 * i + hi16 + r;
        int col = 16 * j + lo16;
        sQ[row * 32 + col] = qd[i][j][r];
        sK[row * 32 + col] = kd[i][j][r];
        hv[r] = (_Float16)vd[i][j][r];
      }
      *(h8v*)(sVT + (16 * j + lo16) * 32 + 16 * i + hi16) = hv;
    }

  // ---- per-head attention
  for (int h = 0; h < 8; ++h) {
    // S^T = k @ (t*q)^T via f32 WMMA K=4 (exact head_dim); temperature folded into B operand
    const int dOff = 4 * h + ((lane & 16) ? 2 : 0);
    const float t = sTemp[h];
    f2v aK[2], bQ[2];
#pragma unroll
    for (int i = 0; i < 2; ++i) aK[i] = *(const f2v*)(sK + (16 * i + lo16) * 32 + dOff);
#pragma unroll
    for (int j = 0; j < 2; ++j) {
      bQ[j] = *(const f2v*)(sQ + (16 * j + lo16) * 32 + dOff);
      bQ[j][0] *= t; bQ[j][1] *= t;
    }

    f8v st[2][2];
#pragma unroll
    for (int i = 0; i < 2; ++i)
#pragma unroll
      for (int j = 0; j < 2; ++j)
        st[i][j] = wmma_f32k4(aK[i], bQ[j], zero);

    // softmax over m: lane n and lane n^16 jointly hold one column l of S^T
#pragma unroll
    for (int j = 0; j < 2; ++j) {
      float mx = -3.0e38f;
#pragma unroll
      for (int i = 0; i < 2; ++i)
#pragma unroll
        for (int r = 0; r < 8; ++r) mx = fmaxf(mx, st[i][j][r]);
      mx = fmaxf(mx, fxor16(mx));
      float sum = 0.0f;
#pragma unroll
      for (int i = 0; i < 2; ++i)
#pragma unroll
        for (int r = 0; r < 8; ++r) {
          float ev = __expf(st[i][j][r] - mx);
          st[i][j][r] = ev;
          sum += ev;
        }
      sum += fxor16(sum);
      float inv = 1.0f / sum;
      // store P row-major [l][m]: contiguous 8 halves per tile
#pragma unroll
      for (int i = 0; i < 2; ++i) {
        h8v hp;
#pragma unroll
        for (int r = 0; r < 8; ++r) hp[r] = (_Float16)(st[i][j][r] * inv);
        *(h8v*)(sP + (16 * j + lo16) * 32 + 16 * i + hi16) = hp;
      }
    }

    // O^T = V^T @ P^T: A rows = head channels (4 valid of 16), B cols = l, K=32 in one WMMA
    int vrow = 4 * h + lo16; if (vrow > 31) vrow = 31;  // padded rows: garbage, never stored
    h16v aV = loadA16(sVT, vrow, lane);
#pragma unroll
    for (int j = 0; j < 2; ++j) {
      h16v bP = loadB16(sP, 16 * j, lane);
      f8v o = wmma_f16(aV, bP, zero);
      if (lane < 16) {  // slots 0..3 of lower half-wave hold the 4 valid head channels
        h4v ho;
#pragma unroll
        for (int r = 0; r < 4; ++r) ho[r] = (_Float16)o[r];
        *(h4v*)(sO + (16 * j + lane) * 32 + 4 * h) = ho;
      }
    }
  }

  // ---- final: F = O@Wo + X@Wr (chained accumulators), then LayerNorm over channels
  f8v fo[2][2];
#pragma unroll
  for (int i = 0; i < 2; ++i) {
    h16v aOv = loadA16(sO, 16 * i + lo16, lane);
    h16v aXv = loadA16(sX, 16 * i + lo16, lane);
#pragma unroll
    for (int j = 0; j < 2; ++j) {
      f8v acc = wmma_f16(aXv, loadB16(sWrT, 16 * j, lane), zero);
      fo[i][j] = wmma_f16(aOv, loadB16(sWoT, 16 * j, lane), acc);
    }
  }

  const float g0 = sGamma[lo16], g1 = sGamma[16 + lo16];
  const float b0 = sBeta[lo16],  b1 = sBeta[16 + lo16];
  float* orow = out + (size_t)e * 1024;
#pragma unroll
  for (int i = 0; i < 2; ++i)
#pragma unroll
    for (int r = 0; r < 8; ++r) {
      float a0 = fo[i][0][r], a1 = fo[i][1][r];
      float s1 = a0 + a1, s2 = a0 * a0 + a1 * a1;
      s1 += fxor_dpp<DPP_XOR1>(s1); s2 += fxor_dpp<DPP_XOR1>(s2);
      s1 += fxor_dpp<DPP_XOR2>(s1); s2 += fxor_dpp<DPP_XOR2>(s2);
      s1 += fxor_dpp<DPP_XOR4>(s1); s2 += fxor_dpp<DPP_XOR4>(s2);
      s1 += fxor_dpp<DPP_XOR8>(s1); s2 += fxor_dpp<DPP_XOR8>(s2);
      float mu = s1 * 0.03125f;
      float var = s2 * 0.03125f - mu * mu;
      float rstd = rsqrtf(var + 1e-5f);
      int row = 16 * i + hi16 + r;
      __builtin_nontemporal_store((a0 - mu) * rstd * g0 + b0, orow + row * 32 + lo16);
      __builtin_nontemporal_store((a1 - mu) * rstd * g1 + b1, orow + row * 32 + 16 + lo16);
    }
}

extern "C" void kernel_launch(void* const* d_in, const int* in_sizes, int n_in,
                              void* d_out, int out_size, void* d_ws, size_t ws_size,
                              hipStream_t stream) {
  const float* x     = (const float*)d_in[0];
  const float* y     = (const float*)d_in[1];
  const float* Wq    = (const float*)d_in[2];
  const float* Wkv   = (const float*)d_in[3];
  const float* Wo    = (const float*)d_in[4];
  const float* Wr    = (const float*)d_in[5];
  const float* temp  = (const float*)d_in[6];
  const float* gamma = (const float*)d_in[7];
  const float* beta  = (const float*)d_in[8];
  float* outp = (float*)d_out;

  int nb = in_sizes[0] / 1024;  // b = elements / (L*C)
  int blocks = (nb + WAVES_PER_BLOCK - 1) / WAVES_PER_BLOCK;
  XMRCA_attn_ln_kernel<<<blocks, BLOCK_THREADS, SMEM_TOTAL, stream>>>(
      x, y, Wq, Wkv, Wo, Wr, temp, gamma, beta, outp, nb);
}
